// LSTMForecaster_46514495815793
// MI455X (gfx1250) — matrix operand order repaired and checked by reading, not compiled, not run on hardware
//
#include <hip/hip_runtime.h>
#include <math.h>

typedef __bf16 v16bf __attribute__((ext_vector_type(16)));
typedef float  v8f   __attribute__((ext_vector_type(8)));
typedef unsigned short u16;

#define HID   1024
#define BCH   128
#define LOOKB 256
#define HORZ  24

union AFrag {
    uint4 u[2];
    v16bf v;
};

__device__ __forceinline__ u16 f2bf_rne(float f) {
    unsigned int u = __float_as_uint(f);
    unsigned int r = u + 0x7FFFu + ((u >> 16) & 1u);
    return (u16)(r >> 16);
}

__device__ __forceinline__ float sigmoidf_(float z) {
    return 1.0f / (1.0f + __expf(-z));
}

// ---------------------------------------------------------------------------
// Fused LSTM cell step: one wave owns a 16x16 (m,j) tile and accumulates all
// four gate tiles (i,f,g,o) with v_wmma_f32_16x16x32_bf16, then applies the
// cell nonlinearity in-register. Grid: 64 blocks x 256 thr = 512 waves =
// (128/16) m-tiles x (1024/16) j-tiles.
// ---------------------------------------------------------------------------
__global__ __launch_bounds__(256) void lstm_step_kernel(
    const u16*   __restrict__ x_bf,   // B x HID bf16 input (big-K path) or null
    int ldx,
    const float* __restrict__ x_f,    // B x kxs f32 input (small-K path) or null
    int kxs, int ldxf,
    const float* __restrict__ wih_f,  // 4H x kxs f32 (small-K path)
    const u16*   __restrict__ wih_bf, // 4H x HID bf16 (big-K path)
    const u16*   __restrict__ whh_bf, // 4H x HID bf16
    const float* __restrict__ bias,   // 4H
    const u16*   __restrict__ h_in,   // B x HID bf16
    float*       __restrict__ cst,    // B x HID f32, updated in place
    u16*         __restrict__ h_out_bf, // B x HID bf16
    float*       __restrict__ h_out_f)  // B x HID f32 or null
{
    const int lane = threadIdx.x & 31;
    const int wave = threadIdx.x >> 5;
    const int tile = blockIdx.x * (blockDim.x >> 5) + wave;
    const int JT   = HID / 16;                       // 64
    if (tile >= (BCH / 16) * JT) return;             // wave-uniform
    const int m_base = (tile / JT) * 16;
    const int j_base = (tile % JT) * 16;
    const int lcol = lane & 15;
    const int hi   = lane >> 4;

    v8f acc[4];
    #pragma unroll
    for (int g = 0; g < 4; ++g)
        #pragma unroll
        for (int r = 0; r < 8; ++r) acc[g][r] = 0.0f;

    // ---- recurrent GEMM: h_in (16 x 1024) x Whh^T tiles, K chunks of 32 ----
    {
        const u16* aptr = h_in + (size_t)(m_base + lcol) * HID;
        for (int k0 = 0; k0 < HID; k0 += 32) {
            AFrag a;
            const uint4* pa = reinterpret_cast<const uint4*>(aptr + k0 + (hi ? 16 : 0));
            a.u[0] = pa[0];        // K {0..7} (+16 if hi)
            a.u[1] = pa[1];        // K {8..15} (+16 if hi)
            #pragma unroll
            for (int g = 0; g < 4; ++g) {
                AFrag b;
                const uint4* pb = reinterpret_cast<const uint4*>(
                    whh_bf + (size_t)(g * HID + j_base + lcol) * HID + k0 + (hi ? 16 : 0));
                b.u[0] = pb[0];
                b.u[1] = pb[1];
                acc[g] = __builtin_amdgcn_wmma_f32_16x16x32_bf16(
                    false, a.v, false, b.v, (short)0, acc[g], false, false);
            }
        }
    }

    // ---- input GEMM (big-K, bf16 path): x (16 x 1024) x Wih^T tiles ----
    if (x_bf) {
        const u16* aptr = x_bf + (size_t)(m_base + lcol) * ldx;
        for (int k0 = 0; k0 < HID; k0 += 32) {
            AFrag a;
            const uint4* pa = reinterpret_cast<const uint4*>(aptr + k0 + (hi ? 16 : 0));
            a.u[0] = pa[0];
            a.u[1] = pa[1];
            #pragma unroll
            for (int g = 0; g < 4; ++g) {
                AFrag b;
                const uint4* pb = reinterpret_cast<const uint4*>(
                    wih_bf + (size_t)(g * HID + j_base + lcol) * HID + k0 + (hi ? 16 : 0));
                b.u[0] = pb[0];
                b.u[1] = pb[1];
                acc[g] = __builtin_amdgcn_wmma_f32_16x16x32_bf16(
                    false, a.v, false, b.v, (short)0, acc[g], false, false);
            }
        }
    }

    // ---- epilogue: bias + small-K f32 input path + LSTM nonlinearity ----
    const int j = j_base + lcol;
    const float bi = bias[0 * HID + j];
    const float bf = bias[1 * HID + j];
    const float bg = bias[2 * HID + j];
    const float bo = bias[3 * HID + j];

    #pragma unroll
    for (int r = 0; r < 8; ++r) {
        const int m = m_base + r + (hi ? 8 : 0);
        float gi = acc[0][r] + bi;
        float gf = acc[1][r] + bf;
        float gg = acc[2][r] + bg;
        float go = acc[3][r] + bo;
        if (x_f) {
            for (int k = 0; k < kxs; ++k) {
                const float xv = x_f[(size_t)m * ldxf + k];
                gi += xv * wih_f[(size_t)(0 * HID + j) * kxs + k];
                gf += xv * wih_f[(size_t)(1 * HID + j) * kxs + k];
                gg += xv * wih_f[(size_t)(2 * HID + j) * kxs + k];
                go += xv * wih_f[(size_t)(3 * HID + j) * kxs + k];
            }
        }
        const float ig = sigmoidf_(gi);
        const float fg = sigmoidf_(gf);
        const float gt = tanhf(gg);
        const float og = sigmoidf_(go);
        const size_t idx = (size_t)m * HID + j;
        const float cn = fg * cst[idx] + ig * gt;
        cst[idx] = cn;
        const float hn = og * tanhf(cn);
        h_out_bf[idx] = f2bf_rne(hn);
        if (h_out_f) h_out_f[idx] = hn;
    }
}

// ---------------------------------------------------------------------------
__global__ void f32_to_bf16_kernel(const float* __restrict__ src,
                                   u16* __restrict__ dst, int n) {
    int i = blockIdx.x * blockDim.x + threadIdx.x;
    if (i < n) dst[i] = f2bf_rne(src[i]);
}

__global__ void zero_kernel(unsigned int* __restrict__ p, int n32) {
    int i = blockIdx.x * blockDim.x + threadIdx.x;
    if (i < n32) p[i] = 0u;
}

__global__ void seed_decin_kernel(const float* __restrict__ x,
                                  float* __restrict__ dec_in) {
    int m = blockIdx.x * blockDim.x + threadIdx.x;
    if (m < BCH) dec_in[m] = x[(size_t)m * LOOKB * 8 + (size_t)(LOOKB - 1) * 8];
}

// pred[m] = dot(h1_f32[m,:], projW) + pb; writes output & next decoder input
__global__ __launch_bounds__(256) void proj_kernel(
    const float* __restrict__ h1, const float* __restrict__ W,
    const float* __restrict__ pb, float* __restrict__ out,
    float* __restrict__ dec_in, int t) {
    __shared__ float sm[256];
    const int m = blockIdx.x;
    float s = 0.0f;
    for (int jx = threadIdx.x; jx < HID; jx += 256)
        s += h1[(size_t)m * HID + jx] * W[jx];
    sm[threadIdx.x] = s;
    __syncthreads();
    for (int w = 128; w > 0; w >>= 1) {
        if (threadIdx.x < w) sm[threadIdx.x] += sm[threadIdx.x + w];
        __syncthreads();
    }
    if (threadIdx.x == 0) {
        const float p = sm[0] + pb[0];
        out[(size_t)m * HORZ + t] = p;
        dec_in[m] = p;
    }
}

// ---------------------------------------------------------------------------
extern "C" void kernel_launch(void* const* d_in, const int* in_sizes, int n_in,
                              void* d_out, int out_size, void* d_ws, size_t ws_size,
                              hipStream_t stream) {
    const float* x     = (const float*)d_in[0];
    const float* eWih0 = (const float*)d_in[1];
    const float* eWhh0 = (const float*)d_in[2];
    const float* eb0   = (const float*)d_in[3];
    const float* eWih1 = (const float*)d_in[4];
    const float* eWhh1 = (const float*)d_in[5];
    const float* eb1   = (const float*)d_in[6];
    const float* dWih0 = (const float*)d_in[7];
    const float* dWhh0 = (const float*)d_in[8];
    const float* db0   = (const float*)d_in[9];
    const float* dWih1 = (const float*)d_in[10];
    const float* dWhh1 = (const float*)d_in[11];
    const float* db1   = (const float*)d_in[12];
    const float* pW    = (const float*)d_in[13];
    const float* pb    = (const float*)d_in[14];
    float* out = (float*)d_out;

    const size_t BH = (size_t)BCH * HID;     // 131072
    const size_t WN = (size_t)4 * HID * HID; // 4194304 elems per big weight

    char* ws = (char*)d_ws;
    size_t off = 0;
    auto take = [&](size_t bytes) -> char* {
        char* p = ws + off;
        off = (off + bytes + 255) & ~(size_t)255;
        return p;
    };
    u16* wWhh0  = (u16*)take(WN * 2);
    u16* wWih1  = (u16*)take(WN * 2);
    u16* wWhh1  = (u16*)take(WN * 2);
    u16* wdWhh0 = (u16*)take(WN * 2);
    u16* wdWih1 = (u16*)take(WN * 2);
    u16* wdWhh1 = (u16*)take(WN * 2);
    u16* ys0    = (u16*)take((size_t)(LOOKB + 1) * BH * 2); // slice 0 = zero h
    u16* h1p[2]; h1p[0] = (u16*)take(BH * 2); h1p[1] = (u16*)take(BH * 2);
    float* c0 = (float*)take(BH * 4);
    float* c1 = (float*)take(BH * 4);
    u16* dh0p[2]; dh0p[0] = (u16*)take(BH * 2); dh0p[1] = (u16*)take(BH * 2);
    u16* dh1p[2]; dh1p[0] = (u16*)take(BH * 2); dh1p[1] = (u16*)take(BH * 2);
    float* dh1f   = (float*)take(BH * 4);
    float* dec_in = (float*)take(BCH * 4);

    // --- weight precision conversion (f32 -> bf16, RNE) ---
    const int cgrid = (int)((WN + 255) / 256);
    f32_to_bf16_kernel<<<cgrid, 256, 0, stream>>>(eWhh0, wWhh0,  (int)WN);
    f32_to_bf16_kernel<<<cgrid, 256, 0, stream>>>(eWih1, wWih1,  (int)WN);
    f32_to_bf16_kernel<<<cgrid, 256, 0, stream>>>(eWhh1, wWhh1,  (int)WN);
    f32_to_bf16_kernel<<<cgrid, 256, 0, stream>>>(dWhh0, wdWhh0, (int)WN);
    f32_to_bf16_kernel<<<cgrid, 256, 0, stream>>>(dWih1, wdWih1, (int)WN);
    f32_to_bf16_kernel<<<cgrid, 256, 0, stream>>>(dWhh1, wdWhh1, (int)WN);

    // --- zero initial state (every call: harness poisons ws once, and we must
    //     be deterministic across graph replays) ---
    auto zero = [&](void* p, size_t bytes) {
        int n32 = (int)(bytes / 4);
        zero_kernel<<<(n32 + 255) / 256, 256, 0, stream>>>((unsigned int*)p, n32);
    };
    zero(ys0, BH * 2);     // h_{-1} of encoder layer 0
    zero(h1p[0], BH * 2);  // h_{-1} of encoder layer 1
    zero(c0, BH * 4);
    zero(c1, BH * 4);
    seed_decin_kernel<<<1, 128, 0, stream>>>(x, dec_in);

    // --- encoder layer 0: small-K f32 input (K=8), bf16 recurrent WMMA ---
    for (int t = 0; t < LOOKB; ++t) {
        lstm_step_kernel<<<64, 256, 0, stream>>>(
            nullptr, 0,
            x + (size_t)t * 8, 8, LOOKB * 8,
            eWih0, nullptr,
            wWhh0, eb0,
            ys0 + (size_t)t * BH, c0,
            ys0 + (size_t)(t + 1) * BH, nullptr);
    }

    // --- encoder layer 1: both input and recurrent GEMMs in bf16 WMMA ---
    int cur = 0;
    for (int t = 0; t < LOOKB; ++t) {
        lstm_step_kernel<<<64, 256, 0, stream>>>(
            ys0 + (size_t)(t + 1) * BH, HID,
            nullptr, 0, 0,
            nullptr, wWih1,
            wWhh1, eb1,
            h1p[cur], c1,
            h1p[cur ^ 1], nullptr);
        cur ^= 1;
    }
    // final encoder states: layer0 -> ys0 slice LOOKB (+ c0), layer1 -> h1p[cur] (+ c1)

    // --- decoder: 24 autoregressive steps ---
    const u16* dh0_in = ys0 + (size_t)LOOKB * BH;
    const u16* dh1_in = h1p[cur];
    int a0 = 0, a1 = 0;
    for (int t = 0; t < HORZ; ++t) {
        lstm_step_kernel<<<64, 256, 0, stream>>>(      // decoder layer 0 (K=1 input)
            nullptr, 0,
            dec_in, 1, 1,
            dWih0, nullptr,
            wdWhh0, db0,
            dh0_in, c0,
            dh0p[a0], nullptr);
        lstm_step_kernel<<<64, 256, 0, stream>>>(      // decoder layer 1
            dh0p[a0], HID,
            nullptr, 0, 0,
            nullptr, wdWih1,
            wdWhh1, db1,
            dh1_in, c1,
            dh1p[a1], dh1f);
        proj_kernel<<<BCH, 256, 0, stream>>>(dh1f, pW, pb, out, dec_in, t);
        dh0_in = dh0p[a0]; a0 ^= 1;
        dh1_in = dh1p[a1]; a1 ^= 1;
    }
}
